// Attention_12704513261709
// MI455X (gfx1250) — matrix operand order repaired and checked by reading, not compiled, not run on hardware
//
#include <hip/hip_runtime.h>
#include <hip/hip_bf16.h>

typedef unsigned short u16;
typedef __attribute__((ext_vector_type(8)))  float  v8f;
typedef __attribute__((ext_vector_type(16))) __bf16 v16bf;
typedef __attribute__((ext_vector_type(8)))  u16    v8u;

struct V16pair { v8u lo; v8u hi; };

__device__ __forceinline__ u16 f2bf(float f) {
    unsigned u = __builtin_bit_cast(unsigned, f);
    unsigned r = u + 0x7FFFu + ((u >> 16) & 1u);   // round-to-nearest-even
    return (u16)(r >> 16);
}
__device__ __forceinline__ float bf2f(u16 h) {
    unsigned u = ((unsigned)h) << 16;
    return __builtin_bit_cast(float, u);
}

// ---------------------------------------------------------------------------
// Generic bf16 WMMA GEMM: C(MxN) = A(MxK) @ B(KxN)
//   A: row-major, contraction contiguous (lda = row stride, elements)
//   B: "column-major over N", contraction contiguous (ldb = per-column stride)
//   Batched over blockIdx.z decomposed as (b = z>>3, h = z&7); offsets in elems.
// Each wave computes a 16x64 tile (4x v_wmma_f32_16x16x32_bf16 per k-step).
// Block = 128 threads = 4 wave32s covering 16 rows x 256 cols.
// MODE: 0 = bf16 store, 1 = bf16 store * 0.125 (Q scale),
//       2 = bf16 store transposed into V_t[b][h][dv][k] (hardcoded dims),
//       3 = f32 store (final output)
// ---------------------------------------------------------------------------
template <int MODE>
__global__ __launch_bounds__(128)
void gemm_bf16_wmma(const u16* __restrict__ A, const u16* __restrict__ B,
                    void* __restrict__ Cv,
                    int K, int lda, int ldb, int ldc,
                    long aB, long aH, long bB, long bH, long cB, long cH)
{
    const int lane   = threadIdx.x & 31;
    const int wave   = threadIdx.x >> 5;
    const int zb     = blockIdx.z >> 3;
    const int zh     = blockIdx.z & 7;
    const int m0     = blockIdx.x * 16;
    const int n0     = blockIdx.y * 256 + wave * 64;
    const int lr     = lane & 15;
    const int hiHalf = lane >> 4;     // upper 16 lanes

    const u16* Ab = A + aB * zb + aH * zh;
    const u16* Bb = B + bB * zb + bH * zh;

    // A-operand: lane lr / lr+16 both hold row (m0+lr); K split contiguously (same order as B):
    // lanes 0-15: K = k0..k0+15; lanes 16-31: K = k0+16..k0+31.
    const u16* aRow = Ab + (long)(m0 + lr) * lda + (hiHalf ? 16 : 0);

    // B-operand: lane lr holds column (n+lr), K = k0..k0+15; lane lr+16: K = k0+16..k0+31.
    const u16* bCol[4];
#pragma unroll
    for (int t = 0; t < 4; ++t)
        bCol[t] = Bb + (long)(n0 + t * 16 + lr) * ldb + (hiHalf ? 16 : 0);

    v8f acc[4];
#pragma unroll
    for (int t = 0; t < 4; ++t)
#pragma unroll
        for (int r = 0; r < 8; ++r) acc[t][r] = 0.0f;

    for (int k0 = 0; k0 < K; k0 += 32) {
        v8u alo = *(const v8u*)(aRow + k0);
        v8u ahi = *(const v8u*)(aRow + k0 + 8);
        v16bf a = __builtin_bit_cast(v16bf, V16pair{alo, ahi});
#pragma unroll
        for (int t = 0; t < 4; ++t) {
            v16bf b = *(const v16bf*)(bCol[t] + k0);
            acc[t] = __builtin_amdgcn_wmma_f32_16x16x32_bf16(
                false, a, false, b, (short)0, acc[t], false, false);
        }
    }

    // Epilogue. D layout: lane = N column (lr), VGPR r = row r (+8 for upper lanes).
    const int rbase = m0 + (hiHalf ? 8 : 0);
#pragma unroll
    for (int t = 0; t < 4; ++t) {
        const int gn = n0 + t * 16 + lr;
#pragma unroll
        for (int r = 0; r < 8; ++r) {
            const int gm = rbase + r;
            float vv = acc[t][r];
            if (MODE == 1) vv *= 0.125f;            // SCALE = DIM_HEAD^-0.5
            if (MODE == 3) {
                float* Cf = (float*)Cv + cB * zb + cH * zh;
                Cf[(long)gm * ldc + gn] = vv;
            } else if (MODE == 2) {
                // rows = (b,k) over 4096, cols = (h,dv) over 16384
                // V_t[b][h][dv][k], all dims 2048/8 hardcoded
                u16* Cc = (u16*)Cv;
                long addr = ((((long)(gm >> 11)) * 8 + (gn >> 11)) * 2048
                             + (gn & 2047)) * 2048 + (gm & 2047);
                Cc[addr] = f2bf(vv);
            } else {
                u16* Cc = (u16*)Cv + cB * zb + cH * zh;
                Cc[(long)gm * ldc + gn] = f2bf(vv);
            }
        }
    }
}

// ---------------------------------------------------------------------------
// LayerNorm (elementwise_affine=False) then * (gamma + 1); f32 in -> bf16 out.
// One block (256 threads) per row of 256.
// ---------------------------------------------------------------------------
__global__ __launch_bounds__(256)
void layernorm_bf16(const float* __restrict__ x, const float* __restrict__ gamma,
                    u16* __restrict__ xn)
{
    __shared__ float red[256];
    const int row = blockIdx.x;
    const int tid = threadIdx.x;
    const float v = x[(long)row * 256 + tid];

    red[tid] = v; __syncthreads();
    for (int s = 128; s > 0; s >>= 1) {
        if (tid < s) red[tid] += red[tid + s];
        __syncthreads();
    }
    const float mean = red[0] * (1.0f / 256.0f);
    __syncthreads();

    const float d = v - mean;
    red[tid] = d * d; __syncthreads();
    for (int s = 128; s > 0; s >>= 1) {
        if (tid < s) red[tid] += red[tid + s];
        __syncthreads();
    }
    const float var = red[0] * (1.0f / 256.0f);
    const float r   = rsqrtf(var + 1e-5f);
    xn[(long)row * 256 + tid] = f2bf(d * r * (gamma[tid] + 1.0f));
}

// ---------------------------------------------------------------------------
// In-place row softmax over bf16 rows of length 2048; block 256 x 8 elems.
// ---------------------------------------------------------------------------
__global__ __launch_bounds__(256)
void softmax_rows(u16* __restrict__ S)
{
    __shared__ float red[256];
    u16* p = S + (long)blockIdx.x * 2048;
    const int tid = threadIdx.x;

    v8u d = *(const v8u*)(p + tid * 8);
    float f[8];
    float mx = -3.0e38f;
#pragma unroll
    for (int i = 0; i < 8; ++i) { f[i] = bf2f(d[i]); mx = fmaxf(mx, f[i]); }

    red[tid] = mx; __syncthreads();
    for (int s = 128; s > 0; s >>= 1) {
        if (tid < s) red[tid] = fmaxf(red[tid], red[tid + s]);
        __syncthreads();
    }
    const float rowmax = red[0];
    __syncthreads();

    float sum = 0.0f;
#pragma unroll
    for (int i = 0; i < 8; ++i) { f[i] = __expf(f[i] - rowmax); sum += f[i]; }
    red[tid] = sum; __syncthreads();
    for (int s = 128; s > 0; s >>= 1) {
        if (tid < s) red[tid] += red[tid + s];
        __syncthreads();
    }
    const float inv = 1.0f / red[0];

    v8u o;
#pragma unroll
    for (int i = 0; i < 8; ++i) o[i] = f2bf(f[i] * inv);
    *(v8u*)(p + tid * 8) = o;
}

// ---------------------------------------------------------------------------
__global__ __launch_bounds__(256)
void f32_to_bf16(const float* __restrict__ src, u16* __restrict__ dst, long n)
{
    long i = (long)blockIdx.x * blockDim.x + threadIdx.x;
    long stride = (long)gridDim.x * blockDim.x;
    for (; i < n; i += stride) dst[i] = f2bf(src[i]);
}

// ---------------------------------------------------------------------------
extern "C" void kernel_launch(void* const* d_in, const int* in_sizes, int n_in,
                              void* d_out, int out_size, void* d_ws, size_t ws_size,
                              hipStream_t stream)
{
    (void)in_sizes; (void)n_in; (void)out_size; (void)ws_size;

    const float* x     = (const float*)d_in[0];  // (2,2048,256)
    const float* gamma = (const float*)d_in[1];  // (256,)
    const float* Wq    = (const float*)d_in[2];  // (16384,256)
    const float* Wk    = (const float*)d_in[3];
    const float* Wv    = (const float*)d_in[4];
    const float* Wo    = (const float*)d_in[5];  // (256,16384)

    // sizes (elements)
    const long ROWS  = 4096;          // b*n
    const long DIM   = 256;
    const long INNER = 16384;         // h*dh
    const long NK    = 2048;          // seq len
    const long WELEM = INNER * DIM;   // 4Mi per projection matrix

    char* w = (char*)d_ws;
    u16* XN  = (u16*)w; w += ROWS * DIM * 2;            //   2 MB
    u16* WQB = (u16*)w; w += WELEM * 2;                 //   8 MB
    u16* WKB = (u16*)w; w += WELEM * 2;                 //   8 MB
    u16* WVB = (u16*)w; w += WELEM * 2;                 //   8 MB
    u16* WOB = (u16*)w; w += WELEM * 2;                 //   8 MB
    u16* QF  = (u16*)w; w += ROWS * INNER * 2;          // 128 MB  [b,n,h,dh] (scaled)
    u16* KF  = (u16*)w; w += ROWS * INNER * 2;          // 128 MB  [b,n,h,dh]
    u16* VT  = (u16*)w; w += ROWS * INNER * 2;          // 128 MB  [b,h,dv,k]
    u16* SP  = (u16*)w; w += 16L * NK * NK * 2;         // 128 MB  [b,h,q,k] -> P in place
    u16* OF  = (u16*)w; w += ROWS * INNER * 2;          // 128 MB  [b,n,h*2048+dv]

    // 1) LayerNorm -> bf16
    layernorm_bf16<<<dim3((unsigned)ROWS), dim3(256), 0, stream>>>(x, gamma, XN);

    // 2) Weights -> bf16
    f32_to_bf16<<<dim3(2048), dim3(256), 0, stream>>>(Wq, WQB, WELEM);
    f32_to_bf16<<<dim3(2048), dim3(256), 0, stream>>>(Wk, WKB, WELEM);
    f32_to_bf16<<<dim3(2048), dim3(256), 0, stream>>>(Wv, WVB, WELEM);
    f32_to_bf16<<<dim3(2048), dim3(256), 0, stream>>>(Wo, WOB, WELEM);

    // 3) Q/K/V projections: (4096 x 16384) = xn(4096x256) @ W^T
    dim3 gQKV(256, 64, 1), blk(128);
    gemm_bf16_wmma<1><<<gQKV, blk, 0, stream>>>(XN, WQB, QF, 256, 256, 256, 16384,
                                                0, 0, 0, 0, 0, 0);
    gemm_bf16_wmma<0><<<gQKV, blk, 0, stream>>>(XN, WKB, KF, 256, 256, 256, 16384,
                                                0, 0, 0, 0, 0, 0);
    gemm_bf16_wmma<2><<<gQKV, blk, 0, stream>>>(XN, WVB, VT, 256, 256, 256, 0,
                                                0, 0, 0, 0, 0, 0);

    // 4) S = Q @ K^T per (b,h): 2048 x 2048, K = 2048 (Q pre-scaled)
    dim3 gS(128, 8, 16);
    gemm_bf16_wmma<0><<<gS, blk, 0, stream>>>(
        QF, KF, SP, 2048, 16384, 16384, 2048,
        /*aB*/ 2048L * 16384, /*aH*/ 2048,
        /*bB*/ 2048L * 16384, /*bH*/ 2048,
        /*cB*/ 8L * 2048 * 2048, /*cH*/ 2048L * 2048);

    // 5) softmax rows in place (16 bh * 2048 rows)
    softmax_rows<<<dim3(32768), dim3(256), 0, stream>>>(SP);

    // 6) O = P @ V per (b,h) -> OF[b,n, h*2048+dv]
    gemm_bf16_wmma<0><<<gS, blk, 0, stream>>>(
        SP, VT, OF, 2048, 2048, 2048, 16384,
        /*aB*/ 8L * 2048 * 2048, /*aH*/ 2048L * 2048,
        /*bB*/ 8L * 2048 * 2048, /*bH*/ 2048L * 2048,
        /*cB*/ 2048L * 16384, /*cH*/ 2048);

    // 7) out(4096x256, f32) = OF(4096x16384) @ Wo^T
    dim3 gO(256, 1, 1);
    gemm_bf16_wmma<3><<<gO, blk, 0, stream>>>(
        OF, WOB, d_out, 16384, 16384, 16384, 256,
        0, 0, 0, 0, 0, 0);
}
